// PointNet2ClsSsg_42030549958919
// MI455X (gfx1250) — matrix operand order repaired and checked by reading, not compiled, not run on hardware
//
#include <hip/hip_runtime.h>
#include <hip/hip_bf16.h>
#include <math.h>

typedef __bf16 bf16;
typedef bf16 v16bf __attribute__((ext_vector_type(16)));
typedef bf16 v8bf  __attribute__((ext_vector_type(8)));
typedef float v8f  __attribute__((ext_vector_type(8)));

#define BN_INV_F 0.9999950000374998f

#define BATCH 32
#define NPTS  16384
#define NFEAT 61
#define S1    32
#define S2    16
#define NS    8
#define CIN1  64
#define CIN2  1027
#define CIN2P 1056
#define RAD2  1.0f

// ---------------------------------------------------------------- FPS
// One block per batch. dist kept in global scratch (64KB/batch exceeds the
// 64KB static-LDS cap once reduction arrays are added); it is L2-resident.
__global__ void fps_kernel(const float* __restrict__ xyz, int N, int npoint,
                           int chmajor, float* __restrict__ distbuf,
                           int* __restrict__ out_idx, float* __restrict__ new_xyz) {
  __shared__ float rv[256];
  __shared__ int   ri[256];
  const int b = blockIdx.x;
  const int tid = threadIdx.x;
  const int T = blockDim.x;
  float* dist = distbuf + (size_t)b * N;
  for (int n = tid; n < N; n += T) dist[n] = 1e10f;
  __syncthreads();
  int far = 0;
  for (int i = 0; i < npoint; ++i) {
    float cx, cy, cz;
    if (chmajor) {
      const float* p = xyz + (size_t)b * 3 * N;
      cx = p[far]; cy = p[N + far]; cz = p[2 * N + far];
    } else {
      const float* p = xyz + ((size_t)b * N + far) * 3;
      cx = p[0]; cy = p[1]; cz = p[2];
    }
    if (tid == 0) {
      out_idx[b * npoint + i] = far;
      float* q = new_xyz + ((size_t)b * npoint + i) * 3;
      q[0] = cx; q[1] = cy; q[2] = cz;
    }
    float bv = -1.0f; int bi = 0;
    for (int n = tid; n < N; n += T) {
      float x, y, z;
      if (chmajor) {
        const float* p = xyz + (size_t)b * 3 * N;
        x = p[n]; y = p[N + n]; z = p[2 * N + n];
      } else {
        const float* p = xyz + ((size_t)b * N + n) * 3;
        x = p[0]; y = p[1]; z = p[2];
      }
      float dx = x - cx, dy = y - cy, dz = z - cz;
      float d = dx * dx + dy * dy + dz * dz;
      float nd = fminf(dist[n], d);
      dist[n] = nd;
      if (nd > bv) { bv = nd; bi = n; }   // strict > keeps first occurrence
    }
    rv[tid] = bv; ri[tid] = bi;
    __syncthreads();
    for (int s = T >> 1; s > 0; s >>= 1) {
      if (tid < s) {
        if (rv[tid + s] > rv[tid] ||
            (rv[tid + s] == rv[tid] && ri[tid + s] < ri[tid])) {
          rv[tid] = rv[tid + s]; ri[tid] = ri[tid + s];
        }
      }
      __syncthreads();
    }
    far = ri[0];
    __syncthreads();
  }
}

// ------------------------------------------------- ball query + group (SA1)
// One wave32 per (b,s). First-8-in-index-order within radius via ballot.
__global__ void group_sa1(const float* __restrict__ xyz,
                          const float* __restrict__ feat,
                          const float* __restrict__ new_xyz,
                          bf16* __restrict__ out) {
  const int b = blockIdx.x / S1;
  const int s = blockIdx.x % S1;
  const int lane = threadIdx.x;
  const float* cq = new_xyz + ((size_t)b * S1 + s) * 3;
  const float cx = cq[0], cy = cq[1], cz = cq[2];
  __shared__ int sel[NS];
  __shared__ int cnt;
  if (lane == 0) cnt = 0;
  __syncthreads();
  const float* px = xyz + (size_t)b * 3 * NPTS;
  for (int n0 = 0; n0 < NPTS; n0 += 32) {
    int n = n0 + lane;
    float dx = px[n] - cx, dy = px[NPTS + n] - cy, dz = px[2 * NPTS + n] - cz;
    float d = dx * dx + dy * dy + dz * dz;
    unsigned m = (unsigned)__ballot(d <= RAD2);
    if (lane == 0) {
      int c = cnt;
      while (m && c < NS) { int bp = __builtin_ctz(m); sel[c++] = n0 + bp; m &= m - 1; }
      cnt = c;
    }
    __syncthreads();
    if (cnt >= NS) break;   // uniform: all lanes read same cnt
    __syncthreads();
  }
  if (lane == 0 && cnt == 0) { sel[0] = 0; cnt = 1; }  // unreachable (centroid d=0)
  __syncthreads();
  const int nsel = cnt;
  for (int kk = 0; kk < NS; ++kk) {
    int idx = sel[(kk < nsel) ? kk : 0];
    bf16* row = out + (((size_t)(b * S1 + s)) * NS + kk) * CIN1;
    for (int c = lane; c < CIN1; c += 32) {
      float v;
      if (c == 0)      v = px[idx] - cx;
      else if (c == 1) v = px[NPTS + idx] - cy;
      else if (c == 2) v = px[2 * NPTS + idx] - cz;
      else             v = feat[(size_t)b * NFEAT * NPTS + (size_t)(c - 3) * NPTS + idx];
      row[c] = (bf16)v;
    }
  }
}

// ------------------------------------------------- ball query + group (SA2)
__global__ void group_sa2(const float* __restrict__ l1x,
                          const float* __restrict__ l1p,
                          const float* __restrict__ new_xyz,
                          bf16* __restrict__ out) {
  const int b = blockIdx.x / S2;
  const int s = blockIdx.x % S2;
  const int lane = threadIdx.x;
  const float* cq = new_xyz + ((size_t)b * S2 + s) * 3;
  const float cx = cq[0], cy = cq[1], cz = cq[2];
  __shared__ int sel[NS];
  __shared__ int cnt;
  const float* p = l1x + ((size_t)b * S1 + lane) * 3;   // N2 = 32 = one chunk
  float dx = p[0] - cx, dy = p[1] - cy, dz = p[2] - cz;
  float d = dx * dx + dy * dy + dz * dz;
  unsigned m = (unsigned)__ballot(d <= RAD2);
  if (lane == 0) {
    int c = 0;
    while (m && c < NS) { int bp = __builtin_ctz(m); sel[c++] = bp; m &= m - 1; }
    if (c == 0) { sel[0] = 0; c = 1; }
    cnt = c;
  }
  __syncthreads();
  const int nsel = cnt;
  for (int kk = 0; kk < NS; ++kk) {
    int idx = sel[(kk < nsel) ? kk : 0];
    const float* q = l1x + ((size_t)b * S1 + idx) * 3;
    const float* f = l1p + ((size_t)b * S1 + idx) * 1024;
    bf16* row = out + (((size_t)(b * S2 + s)) * NS + kk) * CIN2P;
    for (int c = lane; c < CIN2P; c += 32) {
      float v;
      if (c < 3)         v = q[c] - ((c == 0) ? cx : (c == 1) ? cy : cz);
      else if (c < CIN2) v = f[c - 3];
      else               v = 0.0f;
      row[c] = (bf16)v;
    }
  }
}

// ------------------------------------------------- SA3 group_all (no centering)
__global__ void build_sa3(const float* __restrict__ l2x,
                          const float* __restrict__ l2p,
                          bf16* __restrict__ out, int total) {
  int t = blockIdx.x * blockDim.x + threadIdx.x;
  if (t >= total) return;
  int c = t % CIN2P;
  int mrow = t / CIN2P;     // mrow = b*16 + j
  float v;
  if (c < 3)         v = l2x[(size_t)mrow * 3 + c];
  else if (c < CIN2) v = l2p[(size_t)mrow * 1024 + (c - 3)];
  else               v = 0.0f;
  out[t] = (bf16)v;
}

// ------------------------------------------------- weight f32 -> padded bf16
__global__ void wconvert(const float* __restrict__ W, bf16* __restrict__ Wb,
                         int Cin, int CinPad, int total) {
  int t = blockIdx.x * blockDim.x + threadIdx.x;
  if (t >= total) return;
  int c = t % CinPad;
  int o = t / CinPad;
  Wb[t] = (c < Cin) ? (bf16)W[(size_t)o * Cin + c] : (bf16)0.0f;
}

// ------------------------------------------------- WMMA GEMM + bias + BN + ReLU
// One wave per 16x16 output tile.  D = A(X tile) x B(W^T tile) + C.
// A layout (ISA 7.12.2, 16-bit A 16x32): lane<16 row m, K {k0..k0+7,k0+16..k0+23};
// lane>=16 same rows, K shifted by 8.  B: same K mapping as A (column n = lane&15).
// C/D: n = lane&15, element r -> m = r + 8*(lane>=16).
__global__ void __launch_bounds__(32)
wmma_gemm_bn_relu(const bf16* __restrict__ X, const bf16* __restrict__ Wb,
                  const float* __restrict__ bias, const float* __restrict__ gamma,
                  const float* __restrict__ beta, bf16* __restrict__ Y,
                  int CinPad, int Cout) {
  const int tm = blockIdx.x;
  const int tn = blockIdx.y;
  const int lane = threadIdx.x;
  const int half = lane >> 4;
  const int l15 = lane & 15;
  const bf16* xrow = X + ((size_t)tm * 16 + l15) * CinPad;
  const bf16* wrow = Wb + ((size_t)tn * 16 + l15) * CinPad;
  v8f acc = {};
  for (int k0 = 0; k0 < CinPad; k0 += 32) {
    const int ka = k0 + half * 8;
    v8bf alo = *(const v8bf*)(xrow + ka);
    v8bf ahi = *(const v8bf*)(xrow + ka + 16);
    const int kb = k0 + half * 8;
    v8bf blo = *(const v8bf*)(wrow + kb);
    v8bf bhi = *(const v8bf*)(wrow + kb + 16);
    v16bf a, bm;
#pragma unroll
    for (int i = 0; i < 8; ++i) {
      a[i] = alo[i];  a[i + 8] = ahi[i];
      bm[i] = blo[i]; bm[i + 8] = bhi[i];
    }
    acc = __builtin_amdgcn_wmma_f32_16x16x32_bf16(false, a, false, bm,
                                                  (short)0, acc, false, false);
  }
  const int o = tn * 16 + l15;
  const float bb = bias[o];
  const float gs = gamma[o] * BN_INV_F;
  const float be = beta[o];
#pragma unroll
  for (int r = 0; r < 8; ++r) {
    int mrow = tm * 16 + r + half * 8;
    float v = (acc[r] + bb) * gs + be;
    Y[(size_t)mrow * Cout + o] = (bf16)fmaxf(v, 0.0f);
  }
}

// ------------------------------------------------- max over nsample
__global__ void maxpool(const bf16* __restrict__ Yk, float* __restrict__ P,
                        int k, int C) {
  const int g = blockIdx.x;
  for (int c = threadIdx.x; c < C; c += blockDim.x) {
    float mv = -1e30f;
    for (int j = 0; j < k; ++j)
      mv = fmaxf(mv, (float)Yk[((size_t)g * k + j) * C + c]);
    P[(size_t)g * C + c] = mv;
  }
}

// ------------------------------------------------- FC head (f32)
__global__ void fc_bn_relu(const float* __restrict__ X, const float* __restrict__ W,
                           const float* __restrict__ bias, const float* __restrict__ g,
                           const float* __restrict__ be, float* __restrict__ Y,
                           int rows, int Cin, int Cout) {
  int t = blockIdx.x * blockDim.x + threadIdx.x;
  if (t >= rows * Cout) return;
  int r = t / Cout, o = t % Cout;
  const float* x = X + (size_t)r * Cin;
  const float* w = W + (size_t)o * Cin;
  float acc = 0.f;
  for (int c = 0; c < Cin; ++c) acc += x[c] * w[c];
  float v = (acc + bias[o]) * (g[o] * BN_INV_F) + be[o];
  Y[t] = fmaxf(v, 0.f);
}

__global__ void fc_sigmoid(const float* __restrict__ X, const float* __restrict__ W,
                           const float* __restrict__ bias, float* __restrict__ out,
                           int rows, int Cin) {
  int t = blockIdx.x * blockDim.x + threadIdx.x;
  if (t >= rows) return;
  const float* x = X + (size_t)t * Cin;
  float acc = 0.f;
  for (int c = 0; c < Cin; ++c) acc += x[c] * W[c];
  acc += bias[0];
  out[t] = 1.0f / (1.0f + expf(-acc));
}

// =================================================================== host
extern "C" void kernel_launch(void* const* d_in, const int* in_sizes, int n_in,
                              void* d_out, int out_size, void* d_ws, size_t ws_size,
                              hipStream_t stream) {
  (void)in_sizes; (void)n_in; (void)out_size; (void)ws_size;
  const float* xyz  = (const float*)d_in[0];
  const float* feat = (const float*)d_in[1];
  auto PW = [&](int i) { return (const float*)d_in[i]; };
  const int A1 = 2, A2 = 14, A3 = 26;   // param bases: layer l at base + 4*l (W,b,g,be)

  char* base = (char*)d_ws;
  size_t off = 0;
  auto carve = [&](size_t bytes) {
    void* p = base + off;
    off = (off + bytes + 255) & ~(size_t)255;
    return p;
  };
  const int couts[3]  = {256, 512, 1024};
  const int cin1[3]   = {64, 256, 512};
  const int cin23[3]  = {CIN2, 256, 512};
  const int cin23p[3] = {CIN2P, 256, 512};
  bf16* wb1[3]; bf16* wb2[3]; bf16* wb3[3];
  for (int l = 0; l < 3; ++l) wb1[l] = (bf16*)carve((size_t)couts[l] * cin1[l] * 2);
  for (int l = 0; l < 3; ++l) wb2[l] = (bf16*)carve((size_t)couts[l] * cin23p[l] * 2);
  for (int l = 0; l < 3; ++l) wb3[l] = (bf16*)carve((size_t)couts[l] * cin23p[l] * 2);
  float* distbuf = (float*)carve((size_t)BATCH * NPTS * 4);
  int*   idx1 = (int*)carve((size_t)BATCH * S1 * 4);
  float* nx1  = (float*)carve((size_t)BATCH * S1 * 3 * 4);
  int*   idx2 = (int*)carve((size_t)BATCH * S2 * 4);
  float* nx2  = (float*)carve((size_t)BATCH * S2 * 3 * 4);
  bf16*  actA = (bf16*)carve((size_t)8192 * CIN2P * 2);
  bf16*  actB = (bf16*)carve((size_t)8192 * 1024 * 2);
  float* P1 = (float*)carve((size_t)BATCH * S1 * 1024 * 4);
  float* P2 = (float*)carve((size_t)BATCH * S2 * 1024 * 4);
  float* P3 = (float*)carve((size_t)BATCH * 1024 * 4);
  float* H1 = (float*)carve((size_t)BATCH * 512 * 4);
  float* H2 = (float*)carve((size_t)BATCH * 256 * 4);

  // weights -> padded bf16
  for (int l = 0; l < 3; ++l) {
    int tot = couts[l] * cin1[l];
    wconvert<<<(tot + 255) / 256, 256, 0, stream>>>(PW(A1 + 4 * l), wb1[l], cin1[l], cin1[l], tot);
    tot = couts[l] * cin23p[l];
    wconvert<<<(tot + 255) / 256, 256, 0, stream>>>(PW(A2 + 4 * l), wb2[l], cin23[l], cin23p[l], tot);
    wconvert<<<(tot + 255) / 256, 256, 0, stream>>>(PW(A3 + 4 * l), wb3[l], cin23[l], cin23p[l], tot);
  }

  // -------- SA1: fps -> group -> 3x GEMM -> maxpool
  fps_kernel<<<BATCH, 256, 0, stream>>>(xyz, NPTS, S1, 1, distbuf, idx1, nx1);
  group_sa1<<<BATCH * S1, 32, 0, stream>>>(xyz, feat, nx1, actA);
  wmma_gemm_bn_relu<<<dim3(8192 / 16, 256 / 16), 32, 0, stream>>>(
      actA, wb1[0], PW(A1 + 1), PW(A1 + 2), PW(A1 + 3), actB, 64, 256);
  wmma_gemm_bn_relu<<<dim3(8192 / 16, 512 / 16), 32, 0, stream>>>(
      actB, wb1[1], PW(A1 + 5), PW(A1 + 6), PW(A1 + 7), actA, 256, 512);
  wmma_gemm_bn_relu<<<dim3(8192 / 16, 1024 / 16), 32, 0, stream>>>(
      actA, wb1[2], PW(A1 + 9), PW(A1 + 10), PW(A1 + 11), actB, 512, 1024);
  maxpool<<<BATCH * S1, 256, 0, stream>>>(actB, P1, NS, 1024);

  // -------- SA2
  fps_kernel<<<BATCH, 256, 0, stream>>>(nx1, S1, S2, 0, distbuf, idx2, nx2);
  group_sa2<<<BATCH * S2, 32, 0, stream>>>(nx1, P1, nx2, actA);
  wmma_gemm_bn_relu<<<dim3(4096 / 16, 256 / 16), 32, 0, stream>>>(
      actA, wb2[0], PW(A2 + 1), PW(A2 + 2), PW(A2 + 3), actB, CIN2P, 256);
  wmma_gemm_bn_relu<<<dim3(4096 / 16, 512 / 16), 32, 0, stream>>>(
      actB, wb2[1], PW(A2 + 5), PW(A2 + 6), PW(A2 + 7), actA, 256, 512);
  wmma_gemm_bn_relu<<<dim3(4096 / 16, 1024 / 16), 32, 0, stream>>>(
      actA, wb2[2], PW(A2 + 9), PW(A2 + 10), PW(A2 + 11), actB, 512, 1024);
  maxpool<<<BATCH * S2, 256, 0, stream>>>(actB, P2, NS, 1024);

  // -------- SA3 (group_all)
  {
    int tot = BATCH * 16 * CIN2P;
    build_sa3<<<(tot + 255) / 256, 256, 0, stream>>>(nx2, P2, actA, tot);
  }
  wmma_gemm_bn_relu<<<dim3(512 / 16, 256 / 16), 32, 0, stream>>>(
      actA, wb3[0], PW(A3 + 1), PW(A3 + 2), PW(A3 + 3), actB, CIN2P, 256);
  wmma_gemm_bn_relu<<<dim3(512 / 16, 512 / 16), 32, 0, stream>>>(
      actB, wb3[1], PW(A3 + 5), PW(A3 + 6), PW(A3 + 7), actA, 256, 512);
  wmma_gemm_bn_relu<<<dim3(512 / 16, 1024 / 16), 32, 0, stream>>>(
      actA, wb3[2], PW(A3 + 9), PW(A3 + 10), PW(A3 + 11), actB, 512, 1024);
  maxpool<<<BATCH, 256, 0, stream>>>(actB, P3, 16, 1024);

  // -------- FC head
  fc_bn_relu<<<(BATCH * 512 + 255) / 256, 256, 0, stream>>>(
      P3, PW(38), PW(39), PW(40), PW(41), H1, BATCH, 1024, 512);
  fc_bn_relu<<<(BATCH * 256 + 255) / 256, 256, 0, stream>>>(
      H1, PW(42), PW(43), PW(44), PW(45), H2, BATCH, 512, 256);
  fc_sigmoid<<<1, BATCH, 0, stream>>>(H2, PW(46), PW(47), (float*)d_out, BATCH, 256);
}
